// MultiHead_Attention_Custom_59167469470409
// MI455X (gfx1250) — matrix operand order repaired and checked by reading, not compiled, not run on hardware
//
#include <hip/hip_runtime.h>
#include <hip/hip_bf16.h>

#define BN 4
#define SEQ 2048
#define DIM 1024
#define NH 16
#define DHD 64

typedef __bf16 bf16;
typedef __attribute__((ext_vector_type(16))) __bf16 v16bf;
typedef __attribute__((ext_vector_type(8))) float v8f;

union FragBF { v16bf v; uint4 q[2]; };
union AccF   { v8f  v; float f[8]; };
union V8BF   { uint4 u; bf16 e[8]; };

__device__ __forceinline__ bf16 f2bf(float f) {
  unsigned u = __builtin_bit_cast(unsigned, f);
  unsigned r = (u + 0x7FFFu + ((u >> 16) & 1u)) >> 16;
  unsigned short h = (unsigned short)r;
  return __builtin_bit_cast(bf16, h);
}

// ---- CDNA5 async global->LDS copy (16B per lane), tracked by ASYNCcnt ----
__device__ __forceinline__ void async_copy16(void* lds_dst, const void* gsrc) {
  unsigned lds_addr = (unsigned)(size_t)lds_dst;  // low 32 bits = LDS byte address
  asm volatile("global_load_async_to_lds_b128 %0, %1, off"
               :: "v"(lds_addr), "v"(gsrc) : "memory");
}
__device__ __forceinline__ void async_wait0() {
  asm volatile("s_wait_asynccnt 0x0" ::: "memory");
}

// ---------------- conversion kernels ----------------
__global__ void cvt_kernel(const float* __restrict__ in, bf16* __restrict__ out, int n) {
  int i = blockIdx.x * blockDim.x + threadIdx.x;
  int stride = gridDim.x * blockDim.x;
  for (; i < n; i += stride) out[i] = f2bf(in[i]);
}

// out[n*DIM + k] = in[k*DIM + n]  (transpose W_out so GEMM is A*B^T)
__global__ void cvtT_kernel(const float* __restrict__ in, bf16* __restrict__ out) {
  int i = blockIdx.x * blockDim.x + threadIdx.x;
  if (i < DIM * DIM) {
    int nn = i / DIM, kk = i % DIM;
    out[i] = f2bf(in[kk * DIM + nn]);
  }
}

// ---------------- WMMA GEMM: C[M,N] = A[M,K] * B[N,K]^T ----------------
// Block: 256 threads = 8 waves; block tile 128(M) x 64(N); wave tile 32x32
// (4 accumulators -> 4 WMMAs per K=32 step per wave).
template <bool STORE_F32>
__global__ __launch_bounds__(256) void gemm_nt(const bf16* __restrict__ A,
                                               const bf16* __restrict__ Bm,
                                               bf16* __restrict__ Cb,
                                               float* __restrict__ Cf,
                                               int M, int N, int K) {
  constexpr int LS = 40;  // padded row stride (elements), 80B (16B aligned)
  __shared__ __align__(16) bf16 As[128 * LS];
  __shared__ __align__(16) bf16 Bs[64 * LS];

  const int tid = threadIdx.x;
  const int lane = tid & 31;
  const int wave = tid >> 5;
  const int wm = wave >> 1;        // 0..3 -> M offset wm*32
  const int wn = wave & 1;         // 0..1 -> N offset wn*32
  const int halfsel = lane >> 4;   // 0 | 1
  const int mn = lane & 15;
  const int mbase = blockIdx.y * 128;
  const int nbase = blockIdx.x * 64;

  AccF acc[2][2];
  for (int mi = 0; mi < 2; ++mi)
    for (int ni = 0; ni < 2; ++ni)
      for (int r = 0; r < 8; ++r) acc[mi][ni].f[r] = 0.0f;

  // cooperative staging: A 128x32 (16 el/thread), B 64x32 (8 el/thread)
  const int arow = tid >> 1, acol = (tid & 1) * 16;
  const int brow = tid >> 2, bcol = (tid & 3) * 8;
  const bf16* ag = &A[(size_t)(mbase + arow) * K + acol];
  const bf16* bg = &Bm[(size_t)(nbase + brow) * K + bcol];

  for (int k0 = 0; k0 < K; k0 += 32) {
    __syncthreads();
    async_copy16(&As[arow * LS + acol], ag + k0);
    async_copy16(&As[arow * LS + acol + 8], ag + k0 + 8);
    async_copy16(&Bs[brow * LS + bcol], bg + k0);
    if (k0 + 32 < K) __builtin_prefetch(ag + k0 + 32, 0, 1);
    async_wait0();
    __syncthreads();

    FragBF a[2], b[2];
    for (int mi = 0; mi < 2; ++mi) {
      const bf16* ap = &As[(wm * 32 + mi * 16 + mn) * LS + halfsel * 8];
      a[mi].q[0] = *(const uint4*)ap;
      a[mi].q[1] = *(const uint4*)(ap + 16);
    }
    for (int ni = 0; ni < 2; ++ni) {
      const bf16* bp = &Bs[(wn * 32 + ni * 16 + mn) * LS + halfsel * 8];
      b[ni].q[0] = *(const uint4*)bp;
      b[ni].q[1] = *(const uint4*)(bp + 16);
    }
    for (int mi = 0; mi < 2; ++mi)
      for (int ni = 0; ni < 2; ++ni)
        acc[mi][ni].v = __builtin_amdgcn_wmma_f32_16x16x32_bf16(
            false, a[mi].v, false, b[ni].v, (short)0, acc[mi][ni].v, false, false);
  }

  for (int mi = 0; mi < 2; ++mi)
    for (int ni = 0; ni < 2; ++ni)
      for (int r = 0; r < 8; ++r) {
        int row = mbase + wm * 32 + mi * 16 + halfsel * 8 + r;
        int col = nbase + wn * 32 + ni * 16 + mn;
        if (STORE_F32)
          Cf[(size_t)row * N + col] = acc[mi][ni].f[r];
        else
          Cb[(size_t)row * N + col] = f2bf(acc[mi][ni].f[r]);
      }
}

// ---------------- flash attention: one block per (b, h, 64-query tile) ----------------
__global__ __launch_bounds__(256) void attn_kernel(const bf16* __restrict__ qkv,
                                                   const float* __restrict__ mask,
                                                   bf16* __restrict__ out) {
  constexpr int QS = 72, KS = 72, VS = 40, SS = 36, PS = 40;
  __shared__ __align__(16) bf16 Qs[64 * QS];
  __shared__ __align__(16) bf16 Ks[32 * KS];
  __shared__ __align__(16) bf16 Vt[64 * VS];   // transposed: Vt[d][key]
  __shared__ __align__(16) float Sc[64 * SS];
  __shared__ __align__(16) bf16 Ps[64 * PS];
  __shared__ float corr[64];

  const int tid = threadIdx.x;
  const int lane = tid & 31, wave = tid >> 5;
  const int halfsel = lane >> 4, mn = lane & 15;
  const int qt = blockIdx.x, h = blockIdx.y, b = blockIdx.z;
  const int qbase = qt * 64;
  const size_t rowstride = 3 * DIM;

  // ---- async-load Q tile (64 x 64) ----
  {
    int qr = tid >> 2, qc = (tid & 3) * 16;
    const bf16* src = qkv + ((size_t)b * SEQ + qbase + qr) * rowstride + h * DHD + qc;
    async_copy16(&Qs[qr * QS + qc], src);
    async_copy16(&Qs[qr * QS + qc + 8], src + 8);
  }

  const int qi = wave >> 1;          // query sub-tile 0..3
  const int dblk = (wave & 1) * 32;  // output-dim half 0|32
  AccF oacc[2];
  for (int j = 0; j < 2; ++j)
    for (int r = 0; r < 8; ++r) oacc[j].f[r] = 0.0f;

  float m_i = -3.0e38f, l_i = 0.0f;  // per-row state (valid in tid<64)

  for (int kt = 0; kt < SEQ; kt += 32) {
    // K tile (32 keys x 64 d) via async copy
    {
      int kr = tid >> 3, kc = (tid & 7) * 8;
      const bf16* src =
          qkv + ((size_t)b * SEQ + kt + kr) * rowstride + DIM + h * DHD + kc;
      async_copy16(&Ks[kr * KS + kc], src);
    }
    // V tile, transposed into Vt (needs VGPR round trip)
    {
      int vr = tid >> 3, vc = (tid & 7) * 8;
      const bf16* src =
          qkv + ((size_t)b * SEQ + kt + vr) * rowstride + 2 * DIM + h * DHD + vc;
      V8BF vv;
      vv.u = *(const uint4*)src;
      for (int j = 0; j < 8; ++j) Vt[(vc + j) * VS + vr] = vv.e[j];
    }
    async_wait0();
    __syncthreads();

    // ---- scores = Q * K^T  (wave wm: q sub-tile, wn: key sub-tile) ----
    {
      const int wm = wave >> 1, wn = wave & 1;
      AccF s;
      for (int r = 0; r < 8; ++r) s.f[r] = 0.0f;
      for (int dk = 0; dk < DHD; dk += 32) {
        FragBF a, kb;
        const bf16* ap = &Qs[(wm * 16 + mn) * QS + dk + halfsel * 8];
        a.q[0] = *(const uint4*)ap;
        a.q[1] = *(const uint4*)(ap + 16);
        const bf16* bp = &Ks[(wn * 16 + mn) * KS + dk + halfsel * 8];
        kb.q[0] = *(const uint4*)bp;
        kb.q[1] = *(const uint4*)(bp + 16);
        s.v = __builtin_amdgcn_wmma_f32_16x16x32_bf16(false, a.v, false, kb.v,
                                                      (short)0, s.v, false, false);
      }
      for (int r = 0; r < 8; ++r)
        Sc[(wm * 16 + halfsel * 8 + r) * SS + wn * 16 + mn] = s.f[r];
    }
    __syncthreads();

    // ---- online softmax, one thread per query row ----
    if (tid < 64) {
      float sv[32];
      float mt = -3.0e38f;
      const float* mrow = mask + (size_t)(qbase + tid) * SEQ + kt;
      for (int c = 0; c < 32; ++c) {
        float s = Sc[tid * SS + c] * 0.125f + mrow[c];  // 1/sqrt(64)
        sv[c] = s;
        mt = fmaxf(mt, s);
      }
      float mnew = fmaxf(m_i, mt);
      float cf = __expf(m_i - mnew);
      float ls = 0.0f;
      for (int c = 0; c < 32; ++c) {
        float p = __expf(sv[c] - mnew);
        Ps[tid * PS + c] = f2bf(p);
        ls += p;
      }
      l_i = l_i * cf + ls;
      m_i = mnew;
      corr[tid] = cf;
    }
    __syncthreads();

    // ---- O = O*corr + P * V ----
    {
      FragBF pa;
      const bf16* pp = &Ps[(qi * 16 + mn) * PS + halfsel * 8];
      pa.q[0] = *(const uint4*)pp;
      pa.q[1] = *(const uint4*)(pp + 16);
      for (int j = 0; j < 2; ++j) {
        for (int r = 0; r < 8; ++r)
          oacc[j].f[r] *= corr[qi * 16 + halfsel * 8 + r];
        FragBF vb;
        const bf16* vp = &Vt[(dblk + j * 16 + mn) * VS + halfsel * 8];
        vb.q[0] = *(const uint4*)vp;
        vb.q[1] = *(const uint4*)(vp + 16);
        oacc[j].v = __builtin_amdgcn_wmma_f32_16x16x32_bf16(
            false, pa.v, false, vb.v, (short)0, oacc[j].v, false, false);
      }
    }
    __syncthreads();  // before next iteration overwrites Ks/Vt/Sc/Ps
  }

  if (tid < 64) corr[tid] = 1.0f / l_i;
  __syncthreads();

  for (int j = 0; j < 2; ++j)
    for (int r = 0; r < 8; ++r) {
      int lrow = qi * 16 + halfsel * 8 + r;
      size_t grow = (size_t)b * SEQ + qbase + lrow;
      int col = h * DHD + dblk + j * 16 + mn;
      out[grow * DIM + col] = f2bf(oacc[j].f[r] * corr[lrow]);
    }
}

// ---------------- host launch ----------------
extern "C" void kernel_launch(void* const* d_in, const int* in_sizes, int n_in,
                              void* d_out, int out_size, void* d_ws, size_t ws_size,
                              hipStream_t stream) {
  const float* x    = (const float*)d_in[0];
  const float* mask = (const float*)d_in[1];
  const float* wqkv = (const float*)d_in[2];
  const float* wout = (const float*)d_in[3];
  float* outp = (float*)d_out;

  char* ws = (char*)d_ws;
  size_t off = 0;
  auto alloc = [&](size_t bytes) -> char* {
    char* p = ws + off;
    off = (off + bytes + 255) & ~(size_t)255;
    return p;
  };
  bf16* x_bf    = (bf16*)alloc((size_t)BN * SEQ * DIM * 2);
  bf16* wqkv_bf = (bf16*)alloc((size_t)3 * DIM * DIM * 2);
  bf16* woutT   = (bf16*)alloc((size_t)DIM * DIM * 2);
  bf16* qkv_bf  = (bf16*)alloc((size_t)BN * SEQ * 3 * DIM * 2);
  bf16* attn_bf = (bf16*)alloc((size_t)BN * SEQ * DIM * 2);

  int nx = BN * SEQ * DIM;
  cvt_kernel<<<(nx + 255) / 256, 256, 0, stream>>>(x, x_bf, nx);
  int nw = 3 * DIM * DIM;
  cvt_kernel<<<(nw + 255) / 256, 256, 0, stream>>>(wqkv, wqkv_bf, nw);
  cvtT_kernel<<<(DIM * DIM + 255) / 256, 256, 0, stream>>>(wout, woutT);

  // qkv[b*s, 3D] = x[b*s, D] * Wqkv[3D, D]^T
  dim3 g1(3 * DIM / 64, BN * SEQ / 128);
  gemm_nt<false><<<g1, 256, 0, stream>>>(x_bf, wqkv_bf, qkv_bf, nullptr,
                                         BN * SEQ, 3 * DIM, DIM);

  // flash attention
  dim3 ga(SEQ / 64, NH, BN);
  attn_kernel<<<ga, 256, 0, stream>>>(qkv_bf, mask, attn_bf);

  // out[b*s, D] = attn[b*s, D] * WoutT[D, D]^T  (f32 output)
  dim3 g3(DIM / 64, BN * SEQ / 128);
  gemm_nt<true><<<g3, 256, 0, stream>>>(attn_bf, woutT, nullptr, outp,
                                        BN * SEQ, DIM, DIM);
}
